// GPT_64347200029289
// MI455X (gfx1250) — matrix operand order repaired and checked by reading, not compiled, not run on hardware
//
#include <hip/hip_runtime.h>

// ---- model dims ----
#define Cc   768
#define Hh   12
#define Ll   12
#define Nh   64
#define Bb   4
#define Tt   1024
#define Qq   256
#define Vv   50304
#define GN_EPSF 6.4e-4f   // 1e-5 * 64

typedef __bf16 bf16;
typedef bf16  v16bf __attribute__((ext_vector_type(16)));
typedef bf16  v8bf  __attribute__((ext_vector_type(8)));
typedef float v8f   __attribute__((ext_vector_type(8)));

__device__ __forceinline__ unsigned short f2bf_u(float f) {
  union { float f; unsigned u; } in; in.f = f;
  unsigned u = in.u;
  u += 0x7fffu + ((u >> 16) & 1u);          // round-to-nearest-even
  return (unsigned short)(u >> 16);
}
__device__ __forceinline__ bf16 f2bf(float f) {
  union { unsigned short s; bf16 b; } out; out.s = f2bf_u(f);
  return out.b;
}

// gfx1250 async global->LDS copy (GLOBAL_LOAD_ASYNC_TO_LDS_B128, ASYNCcnt)
__device__ __forceinline__ void async_copy_b128(void* lds_dst, const void* gsrc) {
  unsigned ldsOff = (unsigned)(size_t)lds_dst;      // flat shared addr[31:0] = LDS offset
  asm volatile("global_load_async_to_lds_b128 %0, %1, off"
               :: "v"(ldsOff), "v"(gsrc) : "memory");
}
__device__ __forceinline__ void wait_async0() {
  asm volatile("s_wait_asynccnt 0x0" ::: "memory");
}

// ---------------------------------------------------------------------------
// Tiled GEMM: Y[M,N] = X[M,K] @ W[K,N] (+bias) (+relu^2) (+resid), bf16 WMMA.
// 64x64 tile / block, 128 threads (4 waves), BK=32, f32 accumulate.
// LDS: A row-major (stride 40 bf16, 16B aligned rows), B TRANSPOSED so both
// fragment types load as 2x ds_load_b128 per lane.
// mode: 0 = bias (+resid if non-null); 1 = relu(acc+bias)^2
// ---------------------------------------------------------------------------
__global__ __launch_bounds__(128) void gemm_wmma_bf16(
    const float* __restrict__ X, const float* __restrict__ W,
    const float* __restrict__ bias, const float* __restrict__ resid,
    float* __restrict__ Y, int Md, int Kd, int Nd, int mode)
{
  __shared__ __align__(16) bf16 As [64][40];   // A tile 64x32, row stride 80 B
  __shared__ __align__(16) bf16 BsT[64][40];   // B tile transposed: [n][k]

  const int tid  = threadIdx.x;
  const int wave = tid >> 5;
  const int lane = tid & 31;
  const int mBase = blockIdx.y * 64;
  const int nBase = blockIdx.x * 64;

  // staging assignments (one shot per K-step, all 128 threads)
  const int arow = tid >> 1;            // A: row 0..63
  const int acb  = (tid & 1) << 4;      // A: col 0 or 16
  const int bk2  = (tid >> 3) << 1;     // B: k pair 0,2,..,30
  const int bnb  = (tid & 7) << 3;      // B: n group of 8

  v8f acc[4];
  #pragma unroll
  for (int nt = 0; nt < 4; ++nt)
    #pragma unroll
    for (int e = 0; e < 8; ++e) acc[nt][e] = 0.f;

  for (int k0 = 0; k0 < Kd; k0 += 32) {
    // ---- stage A: 4x b128 global loads -> 2x b128 LDS stores ----
    {
      const float4* gp = (const float4*)(X + (size_t)(mBase + arow) * Kd + k0 + acb);
      float4 q0 = gp[0], q1 = gp[1], q2 = gp[2], q3 = gp[3];
      v8bf p0, p1;
      p0[0]=f2bf(q0.x); p0[1]=f2bf(q0.y); p0[2]=f2bf(q0.z); p0[3]=f2bf(q0.w);
      p0[4]=f2bf(q1.x); p0[5]=f2bf(q1.y); p0[6]=f2bf(q1.z); p0[7]=f2bf(q1.w);
      p1[0]=f2bf(q2.x); p1[1]=f2bf(q2.y); p1[2]=f2bf(q2.z); p1[3]=f2bf(q2.w);
      p1[4]=f2bf(q3.x); p1[5]=f2bf(q3.y); p1[6]=f2bf(q3.z); p1[7]=f2bf(q3.w);
      *(v8bf*)&As[arow][acb]     = p0;
      *(v8bf*)&As[arow][acb + 8] = p1;
    }
    // ---- stage B transposed: 2 rows x 8 cols per thread, pack k-pairs -> b32 stores ----
    {
      const float4* r04 = (const float4*)(W + (size_t)(k0 + bk2)     * Nd + nBase + bnb);
      const float4* r14 = (const float4*)(W + (size_t)(k0 + bk2 + 1) * Nd + nBase + bnb);
      float4 a0 = r04[0], a1 = r04[1], b0 = r14[0], b1 = r14[1];
      float v0a[8] = {a0.x, a0.y, a0.z, a0.w, a1.x, a1.y, a1.z, a1.w};
      float v1a[8] = {b0.x, b0.y, b0.z, b0.w, b1.x, b1.y, b1.z, b1.w};
      #pragma unroll
      for (int j = 0; j < 8; ++j) {
        unsigned pk = (unsigned)f2bf_u(v0a[j]) | ((unsigned)f2bf_u(v1a[j]) << 16);
        *(unsigned*)&BsT[bnb + j][bk2] = pk;   // {k, k+1} adjacent bf16
      }
    }
    __syncthreads();

    // prefetch next K-tile (global_prefetch_b8)
    if (k0 + 32 < Kd) {
      __builtin_prefetch(&X[(size_t)(mBase + arow) * Kd + k0 + 32 + acb], 0, 1);
      __builtin_prefetch(&W[(size_t)(k0 + 32 + bk2) * Nd + nBase + bnb], 0, 1);
    }

    // ---- A fragment: 2x ds_load_b128 (ISA 7.12.2 16-bit A 16x32 layout) ----
    const int ar  = (wave << 4) + (lane & 15);
    const int akb = (lane < 16) ? 0 : 8;
    v8bf alo = *(const v8bf*)&As[ar][akb];
    v8bf ahi = *(const v8bf*)&As[ar][akb + 16];
    v16bf a = __builtin_shufflevector(alo, ahi,
        0,1,2,3,4,5,6,7,8,9,10,11,12,13,14,15);

    // ---- B fragments: 2x ds_load_b128 each from transposed tile ----
    const int bkb = (lane < 16) ? 0 : 8;
    const int bcol = lane & 15;
    #pragma unroll
    for (int nt = 0; nt < 4; ++nt) {
      int col = nt * 16 + bcol;
      v8bf blo = *(const v8bf*)&BsT[col][bkb];
      v8bf bhi = *(const v8bf*)&BsT[col][bkb + 16];
      v16bf bbv = __builtin_shufflevector(blo, bhi,
          0,1,2,3,4,5,6,7,8,9,10,11,12,13,14,15);
      acc[nt] = __builtin_amdgcn_wmma_f32_16x16x32_bf16(
          false, a, false, bbv, (short)0, acc[nt], false, false);
    }
    __syncthreads();
  }

  // C/D layout: vgpr r -> row r (lanes 0-15) or 8+r (lanes 16-31), col = lane&15
  const int rowBase = mBase + (wave << 4) + ((lane < 16) ? 0 : 8);
  const int colIn   = lane & 15;
  #pragma unroll
  for (int nt = 0; nt < 4; ++nt) {
    int col = nBase + nt * 16 + colIn;
    float bv = bias ? bias[col] : 0.f;
    #pragma unroll
    for (int r = 0; r < 8; ++r) {
      int row = rowBase + r;
      float v = acc[nt][r] + bv;
      if (mode == 1) v = (v > 0.f) ? v * v : 0.f;
      if (resid) v += resid[(size_t)row * Nd + col];
      Y[(size_t)row * Nd + col] = v;
    }
  }
}

// ---------------------------------------------------------------------------
// Embedding: x[b,t,:] = wte[idx[b,t]] + wpe[t]
// ---------------------------------------------------------------------------
__global__ void embed_k(const int* __restrict__ idx, const float* __restrict__ wte,
                        const float* __restrict__ wpe, float* __restrict__ x)
{
  long i = (long)blockIdx.x * 256 + threadIdx.x;
  const long STOT = (long)Bb * Tt * Cc;
  if (i >= STOT) return;
  int c = (int)(i % Cc);
  long bt = i / Cc;
  int t = (int)(bt % Tt);
  int tok = idx[bt];
  x[i] = wte[(size_t)tok * Cc + c] + wpe[(size_t)t * Cc + c];
}

// ---------------------------------------------------------------------------
// Row LayerNorm over C=768, one block (256 thr) per row; strided rows
// ---------------------------------------------------------------------------
__global__ __launch_bounds__(256) void layernorm_k(
    const float* __restrict__ x, long ldIn, const float* __restrict__ w,
    const float* __restrict__ b, float* __restrict__ out, long ldOut, float eps)
{
  long row = blockIdx.x;
  const float* xr = x + row * ldIn;
  float* orow = out + row * ldOut;
  int tid = threadIdx.x;
  float v0 = xr[tid], v1 = xr[tid + 256], v2 = xr[tid + 512];
  __shared__ float red[256];
  __shared__ float smu, srs;
  red[tid] = v0 + v1 + v2;
  __syncthreads();
  for (int s = 128; s > 0; s >>= 1) { if (tid < s) red[tid] += red[tid + s]; __syncthreads(); }
  if (tid == 0) smu = red[0] * (1.f / Cc);
  __syncthreads();
  float c0 = v0 - smu, c1 = v1 - smu, c2 = v2 - smu;
  red[tid] = c0 * c0 + c1 * c1 + c2 * c2;
  __syncthreads();
  for (int s = 128; s > 0; s >>= 1) { if (tid < s) red[tid] += red[tid + s]; __syncthreads(); }
  if (tid == 0) srs = rsqrtf(red[0] * (1.f / Cc) + eps);
  __syncthreads();
  orow[tid]       = c0 * srs * w[tid]       + b[tid];
  orow[tid + 256] = c1 * srs * w[tid + 256] + b[tid + 256];
  orow[tid + 512] = c2 * srs * w[tid + 512] + b[tid + 512];
}

// ---------------------------------------------------------------------------
// Token-shift mixes
// ---------------------------------------------------------------------------
__global__ void tmix_mix_k(const float* __restrict__ xln,
                           const float* __restrict__ mk, const float* __restrict__ mv,
                           const float* __restrict__ mr, const float* __restrict__ mg,
                           float* __restrict__ xk, float* __restrict__ xv,
                           float* __restrict__ xr, float* __restrict__ xg)
{
  long i = (long)blockIdx.x * 256 + threadIdx.x;
  const long STOT = (long)Bb * Tt * Cc;
  if (i >= STOT) return;
  int c = (int)(i % Cc);
  int t = (int)((i / Cc) % Tt);
  float cur = xln[i];
  float prev = (t > 0) ? xln[i - Cc] : 0.f;
  float xx = prev - cur;
  xk[i] = cur + xx * mk[c];
  xv[i] = cur + xx * mv[c];
  xr[i] = cur + xx * mr[c];
  xg[i] = cur + xx * mg[c];
}

__global__ void cmix_mix_k(const float* __restrict__ xln,
                           const float* __restrict__ mk, const float* __restrict__ mr,
                           float* __restrict__ xk, float* __restrict__ xr)
{
  long i = (long)blockIdx.x * 256 + threadIdx.x;
  const long STOT = (long)Bb * Tt * Cc;
  if (i >= STOT) return;
  int c = (int)(i % Cc);
  int t = (int)((i / Cc) % Tt);
  float cur = xln[i];
  float prev = (t > 0) ? xln[i - Cc] : 0.f;
  float xx = prev - cur;
  xk[i] = cur + xx * mk[c];
  xr[i] = cur + xx * mr[c];
}

// ---------------------------------------------------------------------------
// Chunked linear-attention scan: one block per (b,h), 256 threads = Q rows.
// k/v chunk tiles staged via async global->LDS b128; state f32 in LDS.
// ---------------------------------------------------------------------------
__global__ __launch_bounds__(256) void tmix_attn_k(
    const float* __restrict__ rb, const float* __restrict__ kb,
    const float* __restrict__ vb, float* __restrict__ yb,
    const float* __restrict__ td, const float* __restrict__ tf)
{
  const int bh = blockIdx.x;
  const int b = bh / Hh, h = bh % Hh;
  const float logw = -__expf(td[h]);      // log(w) = -exp(td)
  const float u = tf[h];

  __shared__ float wpow[Qq + 1];
  __shared__ float state[Nh][Nh + 1];              // [k-dim][v-dim]
  __shared__ __align__(16) float ks[64][Nh + 4];   // stride 68 -> 16B aligned rows
  __shared__ __align__(16) float vs[64][Nh + 4];

  const int tid = threadIdx.x;            // chunk-local row i
  wpow[tid] = __expf(logw * (float)tid);
  if (tid == 0) wpow[Qq] = __expf(logw * (float)Qq);
  for (int e = tid; e < Nh * Nh; e += 256) state[e >> 6][e & 63] = 0.f;
  __syncthreads();

  const size_t base = (size_t)b * Tt * Cc + (size_t)h * Nh;
  float r_i[Nh], y_i[Nh], sacc[16];

  for (int ch = 0; ch < Tt / Qq; ++ch) {
    const int t0 = ch * Qq;
    const float4* rp = (const float4*)(rb + base + (size_t)(t0 + tid) * Cc);
    #pragma unroll
    for (int n4 = 0; n4 < 16; ++n4) {
      float4 q = rp[n4];
      r_i[4*n4] = q.x; r_i[4*n4+1] = q.y; r_i[4*n4+2] = q.z; r_i[4*n4+3] = q.w;
    }

    // y_i = (r_i @ state) * w^i
    #pragma unroll
    for (int n = 0; n < Nh; ++n) y_i[n] = 0.f;
    #pragma unroll
    for (int m = 0; m < Nh; ++m) {
      float rm = r_i[m];
      #pragma unroll
      for (int n = 0; n < Nh; ++n) y_i[n] += rm * state[m][n];
    }
    const float wbi = wpow[tid];
    #pragma unroll
    for (int n = 0; n < Nh; ++n) y_i[n] *= wbi;

    #pragma unroll
    for (int q = 0; q < 16; ++q) sacc[q] = 0.f;

    for (int jb = 0; jb < Qq / 64; ++jb) {
      __syncthreads();
      // async-stage k/v 64-row tiles (4x b128 per array per thread)
      for (int e = tid; e < 64 * 16; e += 256) {
        int j = e >> 4, n4 = (e & 15) << 2;
        size_t go = base + (size_t)(t0 + jb * 64 + j) * Cc + n4;
        async_copy_b128(&ks[j][n4], kb + go);
        async_copy_b128(&vs[j][n4], vb + go);
      }
      wait_async0();
      __syncthreads();

      // within-chunk attention contribution for row i = tid
      for (int jj = 0; jj < 64; ++jj) {
        int j = jb * 64 + jj;
        float d = 0.f;
        #pragma unroll
        for (int n = 0; n < Nh; ++n) d += r_i[n] * ks[jj][n];
        float a;
        if (tid > j)       a = d * wpow[tid - j - 1];
        else if (tid == j) a = d * u;
        else               a = 0.f;
        if (a != 0.f) {
          #pragma unroll
          for (int n = 0; n < Nh; ++n) y_i[n] += a * vs[jj][n];
        }
      }

      // partial state update: entry e = tid*16+q -> (n = e>>6 k-dim, m = e&63 v-dim)
      #pragma unroll
      for (int q = 0; q < 16; ++q) {
        int e = tid * 16 + q;
        int n = e >> 6, m = e & 63;
        float accv = sacc[q];
        for (int jj = 0; jj < 64; ++jj) {
          int j = jb * 64 + jj;
          accv += ks[jj][n] * wpow[Qq - 1 - j] * vs[jj][m];
        }
        sacc[q] = accv;
      }
    }

    float4* yp = (float4*)(yb + base + (size_t)(t0 + tid) * Cc);
    #pragma unroll
    for (int n4 = 0; n4 < 16; ++n4) {
      float4 q;
      q.x = y_i[4*n4]; q.y = y_i[4*n4+1]; q.z = y_i[4*n4+2]; q.w = y_i[4*n4+3];
      yp[n4] = q;
    }
    __syncthreads();

    const float wsq = wpow[Qq];
    #pragma unroll
    for (int q = 0; q < 16; ++q) {
      int e = tid * 16 + q;
      int n = e >> 6, m = e & 63;
      state[n][m] = wsq * state[n][m] + sacc[q];
    }
    __syncthreads();
  }
}

// ---------------------------------------------------------------------------
// Per-(token,head) GroupNorm + silu(g) gate. Block = 64 threads.
// ---------------------------------------------------------------------------
__global__ __launch_bounds__(64) void gn_gate_k(
    const float* __restrict__ y, const float* __restrict__ g,
    const float* __restrict__ gnw, const float* __restrict__ gnb,
    float* __restrict__ out)
{
  int blk = blockIdx.x;
  int bt = blk / Hh, h = blk % Hh;
  int n = threadIdx.x;
  size_t baseTok = (size_t)bt * Cc + (size_t)h * Nh;
  float v = y[baseTok + n];
  __shared__ float red[64];
  __shared__ float smu, svar;
  red[n] = v; __syncthreads();
  for (int s = 32; s > 0; s >>= 1) { if (n < s) red[n] += red[n + s]; __syncthreads(); }
  if (n == 0) smu = red[0] * (1.f / Nh);
  __syncthreads();
  float yc = v - smu;
  red[n] = yc * yc; __syncthreads();
  for (int s = 32; s > 0; s >>= 1) { if (n < s) red[n] += red[n + s]; __syncthreads(); }
  if (n == 0) svar = red[0] * (1.f / Nh);
  __syncthreads();
  float o = yc * rsqrtf(svar + GN_EPSF) * gnw[h * Nh + n] + gnb[h * Nh + n];
  float gv = g[baseTok + n];
  out[baseTok + n] = o * (gv / (1.f + __expf(-gv)));   // silu gate
}

// ---------------------------------------------------------------------------
// cmix output: x += sigmoid(rcg) * hv
// ---------------------------------------------------------------------------
__global__ void cmix_out_k(float* __restrict__ x, const float* __restrict__ rcg,
                           const float* __restrict__ hv)
{
  long i = (long)blockIdx.x * 256 + threadIdx.x;
  const long STOT = (long)Bb * Tt * Cc;
  if (i >= STOT) return;
  float s = 1.f / (1.f + __expf(-rcg[i]));
  x[i] += s * hv[i];
}

// ---------------------------------------------------------------------------
// Logits: out[b, v] = xf[b,:] . wte[v,:]   (xf = 4 LN'ed last-token rows)
// ---------------------------------------------------------------------------
__global__ __launch_bounds__(256) void logits_k(
    const float* __restrict__ xf, const float* __restrict__ wte,
    float* __restrict__ out)
{
  __shared__ float xs[Bb][Cc];
  int tid = threadIdx.x;
  for (int e = tid; e < Bb * Cc; e += 256) xs[e / Cc][e % Cc] = xf[e];
  __syncthreads();
  int v = blockIdx.x * 256 + tid;
  if (v >= Vv) return;
  const float4* wr4 = (const float4*)(wte + (size_t)v * Cc);
  float a0 = 0.f, a1 = 0.f, a2 = 0.f, a3 = 0.f;
  for (int c4 = 0; c4 < Cc / 4; ++c4) {
    float4 q = wr4[c4];
    int c = c4 * 4;
    a0 += q.x * xs[0][c] + q.y * xs[0][c+1] + q.z * xs[0][c+2] + q.w * xs[0][c+3];
    a1 += q.x * xs[1][c] + q.y * xs[1][c+1] + q.z * xs[1][c+2] + q.w * xs[1][c+3];
    a2 += q.x * xs[2][c] + q.y * xs[2][c+1] + q.z * xs[2][c+2] + q.w * xs[2][c+3];
    a3 += q.x * xs[3][c] + q.y * xs[3][c+1] + q.z * xs[3][c+2] + q.w * xs[3][c+3];
  }
  out[0 * Vv + v] = a0; out[1 * Vv + v] = a1;
  out[2 * Vv + v] = a2; out[3 * Vv + v] = a3;
}

// ---------------------------------------------------------------------------
extern "C" void kernel_launch(void* const* d_in, const int* in_sizes, int n_in,
                              void* d_out, int out_size, void* d_ws, size_t ws_size,
                              hipStream_t stream) {
  (void)in_sizes; (void)n_in; (void)out_size; (void)ws_size;

  const int*   idx    = (const int*)  d_in[0];
  const float* wte    = (const float*)d_in[1];
  const float* wpe    = (const float*)d_in[2];
  const float* ln1w   = (const float*)d_in[3];
  const float* ln1b   = (const float*)d_in[4];
  const float* ln2w   = (const float*)d_in[5];
  const float* ln2b   = (const float*)d_in[6];
  const float* mtk    = (const float*)d_in[7];
  const float* mtv    = (const float*)d_in[8];
  const float* mtr    = (const float*)d_in[9];
  const float* mtg    = (const float*)d_in[10];
  const float* tdec   = (const float*)d_in[11];
  const float* tfaa   = (const float*)d_in[12];
  const float* Wr     = (const float*)d_in[13];
  const float* br     = (const float*)d_in[14];
  const float* Wk     = (const float*)d_in[15];
  const float* bk     = (const float*)d_in[16];
  const float* Wv     = (const float*)d_in[17];
  const float* bv     = (const float*)d_in[18];
  const float* Wg     = (const float*)d_in[19];
  const float* bg     = (const float*)d_in[20];
  const float* Wo     = (const float*)d_in[21];
  const float* bo     = (const float*)d_in[22];
  const float* gnw    = (const float*)d_in[23];
  const float* gnb    = (const float*)d_in[24];
  const float* cmk    = (const float*)d_in[25];
  const float* cmr    = (const float*)d_in[26];
  const float* Wck    = (const float*)d_in[27];
  const float* bck    = (const float*)d_in[28];
  const float* Wcv    = (const float*)d_in[29];
  const float* bcv    = (const float*)d_in[30];
  const float* Wcr    = (const float*)d_in[31];
  const float* bcr    = (const float*)d_in[32];
  const float* lnfw   = (const float*)d_in[33];
  const float* lnfb   = (const float*)d_in[34];
  float* outp = (float*)d_out;

  const long S  = (long)Bb * Tt * Cc;     // 3,145,728 floats
  const int  M  = Bb * Tt;                // 4096 rows
  float* ws  = (float*)d_ws;              // 12*S floats (~151 MB) used
  float* x   = ws;
  float* xln = ws + 1 * S;
  float* c1  = ws + 2 * S;   // tmix xk / cmix ck-in
  float* c2  = ws + 3 * S;   // tmix xv / cmix cr-in
  float* c3  = ws + 4 * S;   // tmix xr
  float* c4  = ws + 5 * S;   // tmix xg
  float* d1  = ws + 6 * S;   // r
  float* d2  = ws + 7 * S;   // k
  float* d3  = ws + 8 * S;   // v
  float* d4  = ws + 9 * S;   // g (pre-silu)
  float* e   = ws + 10 * S;  // attn y
  float* f   = ws + 11 * S;  // gn-gated y
  float* hb  = d1;           // [M, 3C] overlays d1..d3 (3S, dead by then)
  float* hv  = e;            // cmix value path
  float* rcg = f;            // cmix receptance gate (pre-sigmoid)

  const int FB = (int)((S + 255) / 256);

  auto gemm = [&](const float* X, const float* W, const float* bias,
                  const float* resid, float* Y, int Md, int Kd, int Nd, int mode) {
    dim3 g(Nd / 64, Md / 64), blk(128);
    gemm_wmma_bf16<<<g, blk, 0, stream>>>(X, W, bias, resid, Y, Md, Kd, Nd, mode);
  };

  embed_k<<<FB, 256, 0, stream>>>(idx, wte, wpe, x);

  for (int l = 0; l < Ll; ++l) {
    const size_t wCC = (size_t)l * Cc * Cc;
    // ---- tmix ----
    layernorm_k<<<M, 256, 0, stream>>>(x, Cc, ln1w + l * Cc, ln1b + l * Cc, xln, Cc, 1e-5f);
    tmix_mix_k<<<FB, 256, 0, stream>>>(xln, mtk + l * Cc, mtv + l * Cc,
                                       mtr + l * Cc, mtg + l * Cc, c1, c2, c3, c4);
    gemm(c3, Wr + wCC, br + l * Cc, nullptr, d1, M, Cc, Cc, 0);   // r
    gemm(c1, Wk + wCC, bk + l * Cc, nullptr, d2, M, Cc, Cc, 0);   // k
    gemm(c2, Wv + wCC, bv + l * Cc, nullptr, d3, M, Cc, Cc, 0);   // v
    gemm(c4, Wg + wCC, bg + l * Cc, nullptr, d4, M, Cc, Cc, 0);   // g
    tmix_attn_k<<<Bb * Hh, 256, 0, stream>>>(d1, d2, d3, e, tdec + l * Hh, tfaa + l * Hh);
    gn_gate_k<<<Bb * Tt * Hh, 64, 0, stream>>>(e, d4, gnw + l * Cc, gnb + l * Cc, f);
    gemm(f, Wo + wCC, bo + l * Cc, x, x, M, Cc, Cc, 0);           // x += y@Wo+bo
    // ---- cmix ----
    layernorm_k<<<M, 256, 0, stream>>>(x, Cc, ln2w + l * Cc, ln2b + l * Cc, xln, Cc, 1e-5f);
    cmix_mix_k<<<FB, 256, 0, stream>>>(xln, cmk + l * Cc, cmr + l * Cc, c1, c2);
    gemm(c1, Wck + (size_t)l * Cc * 3 * Cc, bck + (size_t)l * 3 * Cc,
         nullptr, hb, M, Cc, 3 * Cc, 1);                          // relu^2
    gemm(hb, Wcv + (size_t)l * 3 * Cc * Cc, bcv + l * Cc, nullptr, hv, M, 3 * Cc, Cc, 0);
    gemm(c2, Wcr + wCC, bcr + l * Cc, nullptr, rcg, M, Cc, Cc, 0);
    cmix_out_k<<<FB, 256, 0, stream>>>(x, rcg, hv);
  }

  // final LN on the 4 last-token rows, then logits against tied wte
  layernorm_k<<<Bb, 256, 0, stream>>>(x + (size_t)(Tt - 1) * Cc, (long)Tt * Cc,
                                      lnfw, lnfb, xln, Cc, 1e-5f);
  logits_k<<<(Vv + 255) / 256, 256, 0, stream>>>(xln, wte, outp);
}
